// VanillaModel_5471788335120
// MI455X (gfx1250) — matrix operand order repaired and checked by reading, not compiled, not run on hardware
//
#include <hip/hip_runtime.h>
#include <hip/hip_bf16.h>
#include <cstdint>

// ---------------------------------------------------------------------------
// CDNA5 WMMA plumbing (wave32, v_wmma_f32_16x16x32_f16)
// ---------------------------------------------------------------------------
typedef __attribute__((ext_vector_type(16))) _Float16 v16h;
typedef __attribute__((ext_vector_type(8)))  float    v8f;

#define LEAKY(x) ((x) > 0.f ? (x) : 0.1f * (x))

__device__ __forceinline__ v8f wmma16(v16h a, v16h b, v8f c) {
  return __builtin_amdgcn_wmma_f32_16x16x32_f16(false, a, false, b, (short)0, c,
                                                false, false);
}

__device__ __forceinline__ void lds_fence() {
  asm volatile("s_wait_dscnt 0" ::: "memory");
}

// A fragment (16x32 f16, ISA 7.12.2): lane m(0..15) holds row m, K in
// {kb..kb+7, kb+16..kb+23}; lanes 16..31 hold K {kb+8..15, kb+24..31}.
__device__ __forceinline__ v16h load_a_lds(const _Float16* lds, int ld, int kb,
                                           int lane) {
  const _Float16* row = lds + (lane & 15) * ld + kb + ((lane & 16) ? 8 : 0);
  v16h a;
#pragma unroll
  for (int j = 0; j < 8; ++j) {
    a[j]     = row[j];
    a[j + 8] = row[16 + j];
  }
  return a;
}

// B fragment (32x16 f16): lane n(0..15) holds column nb+n, K in
// {kb..kb+7, kb+16..kb+23}; lanes 16..31 hold K {kb+8..15, kb+24..31}. Converted f32->f16 on the fly.
__device__ __forceinline__ v16h load_b_gl(const float* W, int ldw, int kb,
                                          int nb, int lane) {
  const float* p = W + (size_t)(kb + ((lane & 16) ? 8 : 0)) * ldw + nb + (lane & 15);
  v16h b;
#pragma unroll
  for (int j = 0; j < 8; ++j) {
    b[j]     = (_Float16)p[j * ldw];
    b[j + 8] = (_Float16)p[(16 + j) * ldw];
  }
  return b;
}

// D fragment (16x16 f32): element j -> row (j + (lane<16?0:8)), col lane&15.
__device__ __forceinline__ void store_d_lds(_Float16* lds, int ld, int nb,
                                            v8f d, int lane) {
  _Float16* p = lds + ((lane & 16) ? 8 : 0) * ld + nb + (lane & 15);
#pragma unroll
  for (int j = 0; j < 8; ++j) p[j * ld] = (_Float16)d[j];
}

// y = x + leaky(x@w1 + b1) @ w2 + b2 over a 16x64 row tile.
// x held as 4 D-fragments; lds = per-wave 16x72 f16 staging tile.
__device__ __forceinline__ void resblock16(v8f x[4], _Float16* lds,
                                           const float* w1, const float* b1,
                                           const float* w2, const float* b2,
                                           int lane) {
  const int col = lane & 15;
#pragma unroll
  for (int nt = 0; nt < 4; ++nt) store_d_lds(lds, 72, nt * 16, x[nt], lane);
  lds_fence();
  v8f h[4];
#pragma unroll
  for (int nt = 0; nt < 4; ++nt) {
    v8f acc = {};
#pragma unroll
    for (int kb = 0; kb < 64; kb += 32) {
      v16h a = load_a_lds(lds, 72, kb, lane);
      v16h b = load_b_gl(w1, 64, kb, nt * 16, lane);
      acc = wmma16(a, b, acc);
    }
    float bias = b1[nt * 16 + col];
#pragma unroll
    for (int j = 0; j < 8; ++j) {
      float v = acc[j] + bias;
      h[nt][j] = LEAKY(v);
    }
  }
  lds_fence();
#pragma unroll
  for (int nt = 0; nt < 4; ++nt) store_d_lds(lds, 72, nt * 16, h[nt], lane);
  lds_fence();
#pragma unroll
  for (int nt = 0; nt < 4; ++nt) {
    v8f acc = {};
#pragma unroll
    for (int kb = 0; kb < 64; kb += 32) {
      v16h a = load_a_lds(lds, 72, kb, lane);
      v16h b = load_b_gl(w2, 64, kb, nt * 16, lane);
      acc = wmma16(a, b, acc);
    }
    float bias = b2[nt * 16 + col];
#pragma unroll
    for (int j = 0; j < 8; ++j) x[nt][j] += acc[j] + bias;
  }
}

// ---------------------------------------------------------------------------
// Kernels
// ---------------------------------------------------------------------------
#define H_    64
#define NR_   65536
#define NP_C  131072

__global__ void __launch_bounds__(256) k_init(float* gsum, float* gmax,
                                              int* cnt, int n64, int n) {
  int i = blockIdx.x * blockDim.x + threadIdx.x;
  int stride = gridDim.x * blockDim.x;
  for (int j = i; j < n64; j += stride) {
    gsum[j] = 0.f;
    gmax[j] = -3.402823466e38f;
  }
  for (int j = i; j < n; j += stride) cnt[j] = 0;
}

__global__ void __launch_bounds__(128) k_zero_f(float* p, int n) {
  int i = blockIdx.x * blockDim.x + threadIdx.x;
  if (i < n) p[i] = 0.f;
}

// packet featgen: x0 = pe*hlw + hlb; pkt = resblock(hyper); colsum += pkt;
// hpass = resblock(pass.mlp_m, pkt)   (message features, stored f32)
__global__ void __launch_bounds__(256) k_pkt(
    const float* __restrict__ pe, const float* hlw, const float* hlb,
    const float* w1a, const float* b1a, const float* w2a, const float* b2a,
    const float* w1b, const float* b1b, const float* w2b, const float* b2b,
    float* __restrict__ hpass, float* pkt_colsum, int np) {
  __shared__ _Float16 lds[8][16 * 72];
  __shared__ float colsum[64];
  if (threadIdx.x < 64) colsum[threadIdx.x] = 0.f;
  __syncthreads();
  const int lane = threadIdx.x & 31, wave = threadIdx.x >> 5;
  const int base = (blockIdx.x * 8 + wave) * 16;
  const int col = lane & 15, rb = (lane & 16) ? 8 : 0;
  if (base < np) {
    v8f x[4];
#pragma unroll
    for (int nt = 0; nt < 4; ++nt) {
      int c = nt * 16 + col;
      float w = hlw[c], b = hlb[c];
#pragma unroll
      for (int j = 0; j < 8; ++j) x[nt][j] = pe[base + rb + j] * w + b;
    }
    resblock16(x, &lds[wave][0], w1a, b1a, w2a, b2a, lane);  // -> pkt_feat
#pragma unroll
    for (int nt = 0; nt < 4; ++nt) {
      float s = 0.f;
#pragma unroll
      for (int j = 0; j < 8; ++j) s += x[nt][j];
      atomicAdd(&colsum[nt * 16 + col], s);
    }
    resblock16(x, &lds[wave][0], w1b, b1b, w2b, b2b, lane);  // -> h_pass
#pragma unroll
    for (int nt = 0; nt < 4; ++nt)
#pragma unroll
      for (int j = 0; j < 8; ++j)
        hpass[(size_t)(base + rb + j) * H_ + nt * 16 + col] = x[nt][j];
  }
  __syncthreads();
  if (threadIdx.x < 64) atomicAdd(&pkt_colsum[threadIdx.x], colsum[threadIdx.x]);
}

// router featgen: x0 = re[.,0:6]@lw + lb; node = resblock(node_mlp, x0)
__global__ void __launch_bounds__(256) k_router(
    const float* __restrict__ re, const float* lw, const float* lb,
    const float* w1, const float* b1, const float* w2, const float* b2,
    float* __restrict__ node, int nr) {
  __shared__ _Float16 lds[8][16 * 72];
  const int lane = threadIdx.x & 31, wave = threadIdx.x >> 5;
  const int base = (blockIdx.x * 8 + wave) * 16;
  const int col = lane & 15, rb = (lane & 16) ? 8 : 0;
  if (base >= nr) return;
  v8f x[4];
#pragma unroll
  for (int nt = 0; nt < 4; ++nt) {
    int c = nt * 16 + col;
#pragma unroll
    for (int j = 0; j < 8; ++j) {
      const float* r = re + (size_t)(base + rb + j) * 6;
      float s = lb[c];
#pragma unroll
      for (int k = 0; k < 6; ++k) s += r[k] * lw[k * H_ + c];
      x[nt][j] = s;
    }
  }
  resblock16(x, &lds[wave][0], w1, b1, w2, b2, lane);
#pragma unroll
  for (int nt = 0; nt < 4; ++nt)
#pragma unroll
    for (int j = 0; j < 8; ++j)
      node[(size_t)(base + rb + j) * H_ + nt * 16 + col] = x[nt][j];
}

// generic resblock over rows of `in` -> `out` (for h_conn)
__global__ void __launch_bounds__(256) k_resb(
    const float* __restrict__ in, float* __restrict__ out, int n,
    const float* w1, const float* b1, const float* w2, const float* b2) {
  __shared__ _Float16 lds[8][16 * 72];
  const int lane = threadIdx.x & 31, wave = threadIdx.x >> 5;
  const int base = (blockIdx.x * 8 + wave) * 16;
  const int col = lane & 15, rb = (lane & 16) ? 8 : 0;
  if (base >= n) return;
  v8f x[4];
#pragma unroll
  for (int nt = 0; nt < 4; ++nt)
#pragma unroll
    for (int j = 0; j < 8; ++j)
      x[nt][j] = in[(size_t)(base + rb + j) * H_ + nt * 16 + col];
  resblock16(x, &lds[wave][0], w1, b1, w2, b2, lane);
#pragma unroll
  for (int nt = 0; nt < 4; ++nt)
#pragma unroll
    for (int j = 0; j < 8; ++j)
      out[(size_t)(base + rb + j) * H_ + nt * 16 + col] = x[nt][j];
}

// float atomic max via sign-split int trick (native global_atomic_max/min)
__device__ __forceinline__ void atomicMaxF32(float* addr, float val) {
  if (val >= 0.f)
    atomicMax((int*)addr, __float_as_int(val));
  else
    atomicMin((unsigned int*)addr, (unsigned int)__float_as_int(val));
}

// one wave per edge: gather h[src] (64 f32 as 32 float2), scatter sum/max/cnt
__global__ void __launch_bounds__(256) k_edge(
    const int* __restrict__ src, const int* __restrict__ dst, int E,
    const float* __restrict__ h, float* gsum, float* gmax, int* cnt) {
  int wid = (blockIdx.x * blockDim.x + threadIdx.x) >> 5;
  int lane = threadIdx.x & 31;
  if (wid >= E) return;
  int s = src[wid], d = dst[wid];
  float2 v = ((const float2*)(h + (size_t)s * H_))[lane];
  float* sp = gsum + (size_t)d * H_ + lane * 2;
  atomicAdd(sp, v.x);
  atomicAdd(sp + 1, v.y);
  float* mp = gmax + (size_t)d * H_ + lane * 2;
  atomicMaxF32(mp, v.x);
  atomicMaxF32(mp + 1, v.y);
  if (lane == 0) atomicAdd(cnt + d, 1);
}

// combine: node = node + leaky([sum|max|mean|node] @ W[256,64] + B), in place.
// Optionally accumulate per-column sums of the updated node features.
__global__ void __launch_bounds__(128) k_consume(
    const float* __restrict__ gsum, const float* __restrict__ gmax,
    const int* __restrict__ cnt, float* __restrict__ node,
    const float* W, const float* B, float* node_colsum, int nr) {
  __shared__ _Float16 lds[4][16 * 264];
  __shared__ float colsum[64];
  if (threadIdx.x < 64) colsum[threadIdx.x] = 0.f;
  __syncthreads();
  const int lane = threadIdx.x & 31, wave = threadIdx.x >> 5;
  const int base = (blockIdx.x * 4 + wave) * 16;
  const int col = lane & 15, rb = (lane & 16) ? 8 : 0;
  if (base < nr) {
    _Float16* L = &lds[wave][0];
    for (int idx = lane; idx < 16 * 256; idx += 32) {
      int row = idx >> 8, c = idx & 255;
      size_t r = (size_t)(base + row);
      float v;
      if (c < 64)
        v = gsum[r * H_ + c];
      else if (c < 128)
        v = (cnt[r] > 0) ? gmax[r * H_ + (c - 64)] : 0.f;
      else if (c < 192)
        v = gsum[r * H_ + (c - 128)] / fmaxf((float)cnt[r], 1.f);
      else
        v = node[r * H_ + (c - 192)];
      L[row * 264 + c] = (_Float16)v;
    }
    lds_fence();
    v8f y[4];
#pragma unroll
    for (int nt = 0; nt < 4; ++nt) {
      v8f acc = {};
#pragma unroll
      for (int kb = 0; kb < 256; kb += 32) {
        v16h a = load_a_lds(L, 264, kb, lane);
        v16h b = load_b_gl(W, H_, kb, nt * 16, lane);
        acc = wmma16(a, b, acc);
      }
      float bias = B[nt * 16 + col];
#pragma unroll
      for (int j = 0; j < 8; ++j) {
        size_t o = (size_t)(base + rb + j) * H_ + nt * 16 + col;
        float v = acc[j] + bias;
        y[nt][j] = node[o] + LEAKY(v);
      }
    }
#pragma unroll
    for (int nt = 0; nt < 4; ++nt) {
      float s = 0.f;
#pragma unroll
      for (int j = 0; j < 8; ++j) {
        node[(size_t)(base + rb + j) * H_ + nt * 16 + col] = y[nt][j];
        s += y[nt][j];
      }
      if (node_colsum) atomicAdd(&colsum[nt * 16 + col], s);
    }
  }
  __syncthreads();
  if (node_colsum && threadIdx.x < 64)
    atomicAdd(&node_colsum[threadIdx.x], colsum[threadIdx.x]);
}

// head MLP, single block: embed = [pkt_mean | node_mean] (128) -> 64 -> 64 -> 2
__global__ void __launch_bounds__(128) k_head(
    const float* pkt_colsum, const float* node_colsum, const float* w1,
    const float* b1, const float* w2, const float* b2, const float* w3,
    const float* b3, float* __restrict__ out) {
  __shared__ float emb[128], x1[64], x2[64];
  int t = threadIdx.x;
  emb[t] = (t < 64) ? pkt_colsum[t] * (1.f / (float)NP_C)
                    : node_colsum[t - 64] * (1.f / (float)NR_);
  __syncthreads();
  if (t < 64) {
    float s = b1[t];
    for (int k = 0; k < 128; ++k) s += emb[k] * w1[k * H_ + t];
    x1[t] = LEAKY(s);
  }
  __syncthreads();
  if (t < 64) {
    float s = b2[t];
    for (int k = 0; k < 64; ++k) s += x1[k] * w2[k * H_ + t];
    x2[t] = LEAKY(s);
  }
  __syncthreads();
  if (t < 2) {
    float s = b3[t];
    for (int k = 0; k < 64; ++k) s += x2[k] * w3[k * 2 + t];
    out[t] = s;
  }
}

// ---------------------------------------------------------------------------
// Launch
// ---------------------------------------------------------------------------
extern "C" void kernel_launch(void* const* d_in, const int* in_sizes, int n_in,
                              void* d_out, int out_size, void* d_ws,
                              size_t ws_size, hipStream_t stream) {
  const float* router_embed = (const float*)d_in[0];
  const float* packet_embed = (const float*)d_in[1];
  const int* pass_src = (const int*)d_in[2];
  const int* pass_dst = (const int*)d_in[3];
  const int* conn_src = (const int*)d_in[4];
  const int* conn_dst = (const int*)d_in[5];
  // params pytree: dict keys sorted alphabetically at every level
  const float* conn_lin_r_b = (const float*)d_in[6];
  const float* conn_lin_r_w = (const float*)d_in[7];
  const float* conn_m_b1 = (const float*)d_in[8];
  const float* conn_m_b2 = (const float*)d_in[9];
  const float* conn_m_w1 = (const float*)d_in[10];
  const float* conn_m_w2 = (const float*)d_in[11];
  const float* head_b1 = (const float*)d_in[12];
  const float* head_b2 = (const float*)d_in[13];
  const float* head_b3 = (const float*)d_in[14];
  const float* head_w1 = (const float*)d_in[15];
  const float* head_w2 = (const float*)d_in[16];
  const float* head_w3 = (const float*)d_in[17];
  const float* hyper_lin_b = (const float*)d_in[18];
  const float* hyper_lin_w = (const float*)d_in[19];
  const float* hyper_b1 = (const float*)d_in[20];
  const float* hyper_b2 = (const float*)d_in[21];
  const float* hyper_w1 = (const float*)d_in[22];
  const float* hyper_w2 = (const float*)d_in[23];
  const float* node_lin_b = (const float*)d_in[24];
  const float* node_lin_w = (const float*)d_in[25];
  const float* node_b1 = (const float*)d_in[26];
  const float* node_b2 = (const float*)d_in[27];
  const float* node_w1 = (const float*)d_in[28];
  const float* node_w2 = (const float*)d_in[29];
  const float* pass_lin_r_b = (const float*)d_in[30];
  const float* pass_lin_r_w = (const float*)d_in[31];
  const float* pass_m_b1 = (const float*)d_in[32];
  const float* pass_m_b2 = (const float*)d_in[33];
  const float* pass_m_w1 = (const float*)d_in[34];
  const float* pass_m_w2 = (const float*)d_in[35];

  const int NP = 131072, NR = 65536, EP = 1000000, EC = 1000000;

  // workspace carve (~84.2 MB)
  char* w = (char*)d_ws;
  float* hpass = (float*)w;                            // 33,554,432 B (reused as h_conn)
  float* node  = (float*)(w + 33554432);               // 16,777,216 B
  float* gsum  = (float*)(w + 50331648);               // 16,777,216 B
  float* gmax  = (float*)(w + 67108864);               // 16,777,216 B
  int*   cnt   = (int*)  (w + 83886080);               //    262,144 B
  float* csums = (float*)(w + 84148224);               // 128 floats: [pkt|node]
  float* pkt_colsum = csums;
  float* node_colsum = csums + 64;

  // 1. init accumulators + col-sum scratch
  k_init<<<4096, 256, 0, stream>>>(gsum, gmax, cnt, NR * 64, NR);
  k_zero_f<<<1, 128, 0, stream>>>(csums, 128);

  // 2. packet featgen (+ pkt col-sums) and message features h_pass
  k_pkt<<<NP / 16 / 8, 256, 0, stream>>>(
      packet_embed, hyper_lin_w, hyper_lin_b, hyper_w1, hyper_b1, hyper_w2,
      hyper_b2, pass_m_w1, pass_m_b1, pass_m_w2, pass_m_b2, hpass, pkt_colsum,
      NP);

  // 3. router featgen
  k_router<<<NR / 16 / 8, 256, 0, stream>>>(router_embed, node_lin_w,
                                            node_lin_b, node_w1, node_b1,
                                            node_w2, node_b2, node, NR);

  // 4. 'pass' edges: packet -> router scatter sum/max/count
  k_edge<<<(EP + 7) / 8, 256, 0, stream>>>(pass_src, pass_dst, EP, hpass, gsum,
                                           gmax, cnt);

  // 5. 'pass' combine: node += leaky(cat @ lin_r + b)
  k_consume<<<NR / 16 / 4, 128, 0, stream>>>(gsum, gmax, cnt, node,
                                             pass_lin_r_w, pass_lin_r_b,
                                             nullptr, NR);

  // 6. reset accumulators for 'connect'
  k_init<<<4096, 256, 0, stream>>>(gsum, gmax, cnt, NR * 64, NR);

  // 7. h_conn = resblock(conn.mlp_m, node)   (reuses hpass storage)
  k_resb<<<NR / 16 / 8, 256, 0, stream>>>(node, hpass, NR, conn_m_w1,
                                          conn_m_b1, conn_m_w2, conn_m_b2);

  // 8. 'connect' edges: router -> router
  k_edge<<<(EC + 7) / 8, 256, 0, stream>>>(conn_src, conn_dst, EC, hpass, gsum,
                                           gmax, cnt);

  // 9. 'connect' combine + node col-sums
  k_consume<<<NR / 16 / 4, 128, 0, stream>>>(gsum, gmax, cnt, node,
                                             conn_lin_r_w, conn_lin_r_b,
                                             node_colsum, NR);

  // 10. head MLP -> d_out[2]
  k_head<<<1, 128, 0, stream>>>(pkt_colsum, node_colsum, head_w1, head_b1,
                                head_w2, head_b2, head_w3, head_b3,
                                (float*)d_out);
}
